// SimpleRNN_14370960572808
// MI455X (gfx1250) — matrix operand order repaired and checked by reading, not compiled, not run on hardware
//
#include <hip/hip_runtime.h>

typedef __attribute__((ext_vector_type(16))) _Float16 v16h;
typedef __attribute__((ext_vector_type(8)))  float    v8f;
typedef __attribute__((ext_vector_type(2)))  __fp16   v2fp16;

union HalfPack {
    v16h v;
    unsigned int u[8];
    _Float16 h[16];
};

#if __has_builtin(__builtin_amdgcn_tanhf)
#define HW_TANH 1
__device__ __forceinline__ float fast_tanh(float x) {
    return __builtin_amdgcn_tanhf(x);
}
#elif __has_builtin(__builtin_amdgcn_tanh_f32)
#define HW_TANH 1
__device__ __forceinline__ float fast_tanh(float x) {
    return __builtin_amdgcn_tanh_f32(x);
}
#else
__device__ __forceinline__ float fast_tanh(float x) {
    // clamp keeps the exp2 fallback NaN-free; tanh saturates anyway
    x = fminf(10.0f, fmaxf(-10.0f, x));
    float e = __builtin_amdgcn_exp2f(x * 2.8853900817779268f);
    return (e - 1.0f) * __builtin_amdgcn_rcpf(e + 1.0f);
}
#endif

__device__ __forceinline__ unsigned int pack2h(float a, float b) {
    v2fp16 p = __builtin_amdgcn_cvt_pkrtz(a, b);
    return __builtin_bit_cast(unsigned int, p);
}

__global__ __launch_bounds__(64)
void simple_rnn_wmma_kernel(const float* __restrict__ x,
                            const float* __restrict__ W_ih,
                            const float* __restrict__ W_hh,
                            const float* __restrict__ b_ih,
                            const float* __restrict__ b_hh,
                            const float* __restrict__ W_fc,
                            const float* __restrict__ b_fc,
                            float* __restrict__ out,
                            int Btot, int T)
{
    const int lane = threadIdx.x & 31;
    const int wave = blockIdx.x * (blockDim.x >> 5) + (threadIdx.x >> 5);
    const int bbase = wave * 16;
    if (bbase >= Btot) return;               // uniform per wave: whole wave exits

    const int row = lane & 15;               // batch row (B/C/D) or W_hh row (A)
    const int hi  = lane >> 4;               // 0: lanes 0-15, 1: lanes 16-31

    // ---- A = W_hh with the K-space split trick -------------------------------
    // A[m][k]:  k=0..7   = W_hh[m][0..7]   -> lanes 0-15, V0-3
    //           k=8..15  = W_hh[m][8..15]  -> lanes 16-31, V0-3
    //           k=16..31 = 0               -> V4-7 all zero
    // Consequence: B's K=0..7 rows sit in lanes 0-15 V0-3 (h0..7, produced by
    // lanes 0-15) and B's K=8..15 rows sit in lanes 16-31 V0-3 (h8..15,
    // produced by lanes 16-31).  Every lane feeds back its OWN tanh outputs —
    // no cross-lane shuffle in the recurrence.  B's K=16..31 slots (V4-7)
    // multiply zero A columns, so they only need to stay finite (kept at 0).
    HalfPack A;
#pragma unroll
    for (int j = 0; j < 16; ++j)
        A.h[j] = (j >= 8) ? (_Float16)0.0f : (_Float16)W_hh[row * 16 + hi * 8 + j];

    // per-lane constants for C/D-layout rows h = hi*8 + v
    float wih8[8], bias8[8], wfc8[8];
#pragma unroll
    for (int v = 0; v < 8; ++v) {
        int h = hi * 8 + v;
        wih8[v]  = W_ih[h];                  // W_ih is [16,1]
        bias8[v] = b_ih[h] + b_hh[h];
        wfc8[v]  = W_fc[h];                  // W_fc is [1,16]
    }

    // ---- B operand: only u[0..3] are ever written in the loop ----------------
    HalfPack Bm;
#pragma unroll
    for (int j = 0; j < 8; ++j) Bm.u[j] = 0u; // h0 = 0; u[4..7] stay 0 forever

    const float* xrow = x + (size_t)(bbase + row) * (size_t)T;

    float hcur[8];                            // this lane's 8 f32 hidden comps
#pragma unroll
    for (int v = 0; v < 8; ++v) hcur[v] = 0.0f;

    float4 xq = *(const float4*)(xrow);       // t = 0..3 (double buffered)

    for (int t0 = 0; t0 < T; t0 += 4) {
        const int tn = (t0 + 4 < T) ? (t0 + 4) : 0;
        float4 xn = *(const float4*)(xrow + tn);
        float xs[4] = {xq.x, xq.y, xq.z, xq.w};

#pragma unroll
        for (int s = 0; s < 4; ++s) {
            // C = input projection xp[b][t][h] in exact f32 (C-layout)
            v8f c;
#pragma unroll
            for (int v = 0; v < 8; ++v) c[v] = fmaf(xs[s], wih8[v], bias8[v]);

            // D = W_hh x H^T + xp   (f32 accumulate, one matrix op per step)
            v8f d = __builtin_amdgcn_wmma_f32_16x16x32_f16(
                false, A.v, false, Bm.v, (short)0, c, false, false);

            // tanh + pack; each lane feeds back its own 8 components
#pragma unroll
            for (int j = 0; j < 4; ++j) {
                float h0 = fast_tanh(d[2 * j]);
                float h1 = fast_tanh(d[2 * j + 1]);
                hcur[2 * j]     = h0;
                hcur[2 * j + 1] = h1;
                Bm.u[j] = pack2h(h0, h1);
            }
        }
        xq = xn;
    }

    // ---- fc: out[b] = h_T . W_fc + b_fc ----
    float part = 0.0f;
#pragma unroll
    for (int v = 0; v < 8; ++v) part = fmaf(hcur[v], wfc8[v], part);
    float other = __shfl_xor(part, 16, 32);   // one-time cross-half reduce
    if (hi == 0 && (bbase + row) < Btot)
        out[bbase + row] = part + other + b_fc[0];
}

extern "C" void kernel_launch(void* const* d_in, const int* in_sizes, int n_in,
                              void* d_out, int out_size, void* d_ws, size_t ws_size,
                              hipStream_t stream) {
    const float* x    = (const float*)d_in[0];
    const float* W_ih = (const float*)d_in[1];
    const float* W_hh = (const float*)d_in[2];
    const float* b_ih = (const float*)d_in[3];
    const float* b_hh = (const float*)d_in[4];
    const float* W_fc = (const float*)d_in[5];
    const float* b_fc = (const float*)d_in[6];
    float* out = (float*)d_out;

    const int T = 512;
    const int Btot = in_sizes[0] / T;         // 4096

    const int WAVES_PER_BLOCK = 2;            // 64 threads/block -> 128 blocks
    int waves  = (Btot + 15) / 16;
    int blocks = (waves + WAVES_PER_BLOCK - 1) / WAVES_PER_BLOCK;

    simple_rnn_wmma_kernel<<<blocks, WAVES_PER_BLOCK * 32, 0, stream>>>(
        x, W_ih, W_hh, b_ih, b_hh, W_fc, b_fc, out, Btot, T);
}
